// GaussianNewLifterOnline_34394098107051
// MI455X (gfx1250) — matrix operand order repaired and checked by reading, not compiled, not run on hardware
//
#include <hip/hip_runtime.h>

// ---------------------------------------------------------------------------
// MI455X (gfx1250) Gaussian lifter:
//   scalar phase  : per-lane camera transform / masks / anchor / outputs
//   WMMA phase    : f16 16x16x32 WMMA for  df->64->128->96 MLP  +  anchor@W_inst
// Roofline: ~550 MB traffic -> ~23 us @ 23.3 TB/s; 36.5 GFLOP folded into
// v_wmma_f32_16x16x32_f16 (16 KFLOP/instr/wave) so the kernel is HBM-bound.
// ---------------------------------------------------------------------------

typedef _Float16 v16h __attribute__((ext_vector_type(16)));
typedef _Float16 v8h  __attribute__((ext_vector_type(8)));
typedef float    v8f  __attribute__((ext_vector_type(8)));

#define N_PTS 800000
#define BLOCK 256
#define WAVES 8

// LDS layout (units of _Float16).  Row pitches padded +8 f16 (16B) so the
// 16-lane strided ds_load_b128 fragment loads rotate across the 64 banks.
#define WI_PITCH 40    // W_instT rows: k padded 23->32
#define W2_PITCH 72    // 64 + 8
#define W3_PITCH 136   // 128 + 8
#define AN_PITCH 40    // anchor rows: 23 padded ->32, +8
#define H2_PITCH 136
#define OFF_WI   0
#define OFF_W2   (OFF_WI + 96 * WI_PITCH)    // 3840
#define OFF_W3   (OFF_W2 + 128 * W2_PITCH)   // 13056
#define OFF_ACT  (OFF_W3 + 96 * W3_PITCH)    // 26112
#define AN_SIZE  (32 * AN_PITCH)             // 1280
#define H2_SIZE  (32 * H2_PITCH)             // 4352
#define WAVE_ACT (AN_SIZE + H2_SIZE)         // 5632
#define SMEM_F16 (OFF_ACT + WAVES * WAVE_ACT) // 71168 f16 = 142336 B

__device__ __forceinline__ float inv_sig(float t) {
    t = fminf(fmaxf(t, 0.01f), 0.99f);          // subsumes clip(0.001,0.999)
    return logf(t / (1.0f - t));
}

// A-fragment (16x32 f16): lane half splits K in groups of 8 (ISA 7.12.2)
__device__ __forceinline__ v16h load_fragA(const _Float16* rowPtr, int hlf) {
    union { v16h v; v8h h[2]; } u;
    u.h[0] = *(const v8h*)(rowPtr + hlf * 8);
    u.h[1] = *(const v8h*)(rowPtr + 16 + hlf * 8);
    return u.v;
}
// B-fragment (32x16 f16): lane half splits K in groups of 8, same as A
__device__ __forceinline__ v16h load_fragB(const _Float16* rowPtr, int hlf) {
    union { v16h v; v8h h[2]; } u;
    u.h[0] = *(const v8h*)(rowPtr + hlf * 8);
    u.h[1] = *(const v8h*)(rowPtr + 16 + hlf * 8);
    return u.v;
}

__global__ void __launch_bounds__(BLOCK)
lifter_kernel(const float* __restrict__ gp,    const float* __restrict__ zmap,
              const float* __restrict__ w2c,   const float* __restrict__ camk,
              const float* __restrict__ voxo,  const float* __restrict__ ssz,
              const float* __restrict__ cvr,
              const float* __restrict__ Wi,    const float* __restrict__ bi,
              const float* __restrict__ W1,    const float* __restrict__ b1,
              const float* __restrict__ W2,    const float* __restrict__ b2,
              const float* __restrict__ W3,    const float* __restrict__ b3,
              float* __restrict__ out_anchor,  float* __restrict__ out_tag,
              float* __restrict__ out_inst,    float* __restrict__ out_pool,
              float* __restrict__ out_mask) {
    extern __shared__ __align__(16) _Float16 sm[];
    const int tid  = threadIdx.x;
    const int lane = tid & 31;
    const int wave = tid >> 5;
    const int hlf  = lane >> 4;   // 0: lanes 0-15, 1: lanes 16-31
    const int ln16 = lane & 15;

    // ---- stage transposed f16 weight tables in LDS (once per block) ----
    for (int i = tid; i < 96 * 32; i += BLOCK) {
        int n = i >> 5, k = i & 31;
        sm[OFF_WI + n * WI_PITCH + k] = (_Float16)((k < 23) ? Wi[k * 96 + n] : 0.0f);
    }
    for (int i = tid; i < 128 * 64; i += BLOCK) {
        int n = i >> 6, k = i & 63;
        sm[OFF_W2 + n * W2_PITCH + k] = (_Float16)W2[k * 128 + n];
    }
    for (int i = tid; i < 96 * 128; i += BLOCK) {
        int n = i >> 7, k = i & 127;
        sm[OFF_W3 + n * W3_PITCH + k] = (_Float16)W3[k * 96 + n];
    }
    __syncthreads();

    // =================== scalar phase: one point per lane ===================
    const int p = blockIdx.x * BLOCK + tid;      // N = 800000 = 3125 * 256
    const float* g = gp + (size_t)p * 25;
    float gv[25];
#pragma unroll
    for (int i = 0; i < 25; ++i) gv[i] = g[i];

    const float x = gv[0], y = gv[1], zp = gv[2];
    const float camx = w2c[0] * x + w2c[1] * y + w2c[2]  * zp + w2c[3];
    const float camy = w2c[4] * x + w2c[5] * y + w2c[6]  * zp + w2c[7];
    const float camz = w2c[8] * x + w2c[9] * y + w2c[10] * zp + w2c[11];
    const float fx = camk[0], cx0 = camk[2], fy = camk[4], cy0 = camk[5];

    const bool m1 = camz > 1e-6f;
    const float zc = m1 ? camz : 1e-6f;
    const int px = (int)floorf(fx * camx / zc + cx0);
    const int py = (int)floorf(fy * camy / zc + cy0);
    const bool m2 = (px >= 0) && (px < 640) && (py >= 0) && (py < 480);
    const bool mask_all = m1 && m2;

    const float n0 = voxo[0] + 1e-3f, n1 = voxo[1] + 1e-3f, n2 = voxo[2] + 1e-3f;
    const float f0 = voxo[0] + ssz[0] - 1e-3f, f1 = voxo[1] + ssz[1] - 1e-3f,
                f2 = voxo[2] + ssz[2] - 1e-3f;
    const bool gpm = (x > n0) && (x < f0) && (y > n1) && (y < f1) && (zp > n2) && (zp < f2);
    const bool mdet = mask_all && gpm;

#pragma unroll
    for (int i = 0; i < 24; ++i) out_pool[(size_t)p * 25 + i] = gv[i];
    out_pool[(size_t)p * 25 + 24] = gpm ? 1.0f : gv[24];

    float tag = (gv[23] == 1.0f) ? 0.5f : 0.0f;
    if (!mdet) tag = 1.0f;

    const bool gm = (camx >= cvr[0]) && (camx <= cvr[3]) &&
                    (camy >= cvr[1]) && (camy <= cvr[4]) &&
                    (camz >= cvr[2]) && (camz <= cvr[5]);
    const bool reuse = gpm && gm;

    // world2cam quaternion (uniform; cheap per-lane recompute)
    const float tr = w2c[0] + w2c[5] + w2c[10];
    const float qw = sqrtf(fmaxf(tr + 1.0f, 1e-8f)) * 0.5f;
    const float i4 = 1.0f / (4.0f * qw);
    const float qx = (w2c[9] - w2c[6]) * i4;
    const float qy = (w2c[2] - w2c[8]) * i4;
    const float qz = (w2c[4] - w2c[1]) * i4;
    const float p0 = gv[6], p1 = gv[7], p2 = gv[8], p3 = gv[9];
    const float rw = qw * p0 - qx * p1 - qy * p2 - qz * p3;
    const float rx = qw * p1 + qx * p0 + qy * p3 - qz * p2;
    const float ry = qw * p2 - qx * p3 + qy * p0 + qz * p1;
    const float rz = qw * p3 + qx * p2 - qy * p1 + qz * p0;

    const float czs = (fabsf(camz) < 1e-6f) ? 1e-6f : camz;
    const float pixx = fx * camx / czs + cx0;
    const float pixy = fy * camy / czs + cy0;

    float anch[23];
    anch[0] = inv_sig((camx - cvr[0]) / (cvr[3] - cvr[0]));
    anch[1] = inv_sig((camy - cvr[1]) / (cvr[4] - cvr[1]));
    anch[2] = inv_sig((camz - cvr[2]) / (cvr[5] - cvr[2]));
    anch[3] = inv_sig(gv[3]); anch[4] = inv_sig(gv[4]); anch[5] = inv_sig(gv[5]);
    anch[6] = rw; anch[7] = rx; anch[8] = ry; anch[9] = rz;
    anch[10] = inv_sig(gv[10]);
#pragma unroll
    for (int i = 0; i < 12; ++i) anch[11 + i] = gv[11 + i];

    const int ix = (int)fminf(fmaxf(pixx, 0.0f), 639.0f);
    const int iy = (int)fminf(fmaxf(pixy, 0.0f), 479.0f);
    const float df0 = zmap[iy * 640 + ix];   // depth_from_z
    const float df1 = camz;                  // depth_real

    out_mask[p] = reuse ? 1.0f : 0.0f;
    out_tag[p]  = reuse ? tag : 0.0f;
#pragma unroll
    for (int i = 0; i < 23; ++i)
        out_anchor[(size_t)p * 23 + i] = reuse ? anch[i] : 0.0f;

    // stage anchor row (f16, zero-padded K 23..31) into this wave's LDS tile
    _Float16* anbuf = sm + OFF_ACT + wave * WAVE_ACT;
    _Float16* h2buf = anbuf + AN_SIZE;
    {
        _Float16* row = anbuf + lane * AN_PITCH;
#pragma unroll
        for (int i = 0; i < 23; ++i) row[i] = (_Float16)anch[i];
#pragma unroll
        for (int i = 23; i < 32; ++i) row[i] = (_Float16)0.0f;
    }
    const unsigned long long bal = __ballot(reuse);

    // ====================== WMMA phase: 2 M-tiles/wave ======================
    // anchor A-fragments
    v16h aA[2];
#pragma unroll
    for (int mt = 0; mt < 2; ++mt)
        aA[mt] = load_fragA(anbuf + (mt * 16 + ln16) * AN_PITCH, hlf);

    // layer-1 (K=2) straight into A-fragment registers via lane shuffles
    const float d0m0 = __shfl(df0, ln16, 32),      d1m0 = __shfl(df1, ln16, 32);
    const float d0m1 = __shfl(df0, 16 + ln16, 32), d1m1 = __shfl(df1, 16 + ln16, 32);
    v16h h1A[2][2];
#pragma unroll
    for (int ks = 0; ks < 2; ++ks) {
#pragma unroll
        for (int e = 0; e < 16; ++e) {
            const int v = e >> 1, j = e & 1;
            const int kb = (v < 4) ? (hlf * 8 + 2 * v + j)
                                   : (16 + hlf * 8 + 2 * (v - 4) + j);
            const int kk = ks * 32 + kb;
            const float w0 = W1[kk], w1 = W1[64 + kk], bb = b1[kk];
            h1A[0][ks][e] = (_Float16)fmaxf(d0m0 * w0 + d1m0 * w1 + bb, 0.0f);
            h1A[1][ks][e] = (_Float16)fmaxf(d0m1 * w0 + d1m1 * w1 + bb, 0.0f);
        }
    }

    // layer-2: h1(16x64) @ W2(64x128) -> relu -> h2 staged to LDS
#pragma unroll
    for (int ct = 0; ct < 8; ++ct) {
        const _Float16* wr = sm + OFF_W2 + (ct * 16 + ln16) * W2_PITCH;
        const v16h B0 = load_fragB(wr, hlf);
        const v16h B1 = load_fragB(wr + 32, hlf);
        const float b2v = b2[ct * 16 + ln16];
#pragma unroll
        for (int mt = 0; mt < 2; ++mt) {
            v8f acc = {};
            acc = __builtin_amdgcn_wmma_f32_16x16x32_f16(false, h1A[mt][0], false, B0,
                                                         (short)0, acc, false, false);
            acc = __builtin_amdgcn_wmma_f32_16x16x32_f16(false, h1A[mt][1], false, B1,
                                                         (short)0, acc, false, false);
#pragma unroll
            for (int r = 0; r < 8; ++r) {
                const int m = mt * 16 + hlf * 8 + r;       // C layout: M = r + 8*half
                h2buf[m * H2_PITCH + ct * 16 + ln16] =
                    (_Float16)fmaxf(acc[r] + b2v, 0.0f);
            }
        }
    }

    // h2 A-fragments (16x128 -> 4 K-steps per M-tile)
    v16h h2A[2][4];
#pragma unroll
    for (int mt = 0; mt < 2; ++mt)
#pragma unroll
        for (int ks = 0; ks < 4; ++ks)
            h2A[mt][ks] = load_fragA(h2buf + (mt * 16 + ln16) * H2_PITCH + ks * 32, hlf);

    // layer-3 + inst: C = anchor@W_inst + h2@W3, one accumulator chain
    const int pbase = blockIdx.x * BLOCK + wave * 32;
#pragma unroll
    for (int ct = 0; ct < 6; ++ct) {
        const int col = ct * 16 + ln16;
        const v16h BI = load_fragB(sm + OFF_WI + col * WI_PITCH, hlf);
        v16h B3[4];
        const _Float16* w3r = sm + OFF_W3 + col * W3_PITCH;
#pragma unroll
        for (int ks = 0; ks < 4; ++ks) B3[ks] = load_fragB(w3r + ks * 32, hlf);
        const float bias = bi[col] + b3[col];
#pragma unroll
        for (int mt = 0; mt < 2; ++mt) {
            v8f acc = {};
            acc = __builtin_amdgcn_wmma_f32_16x16x32_f16(false, aA[mt], false, BI,
                                                         (short)0, acc, false, false);
#pragma unroll
            for (int ks = 0; ks < 4; ++ks)
                acc = __builtin_amdgcn_wmma_f32_16x16x32_f16(false, h2A[mt][ks], false,
                                                             B3[ks], (short)0, acc,
                                                             false, false);
#pragma unroll
            for (int r = 0; r < 8; ++r) {
                const int pl = mt * 16 + hlf * 8 + r;
                const float vv = ((bal >> pl) & 1ull) ? (acc[r] + bias) : 0.0f;
                out_inst[(size_t)(pbase + pl) * 96 + col] = vv;
            }
        }
    }
}

extern "C" void kernel_launch(void* const* d_in, const int* in_sizes, int n_in,
                              void* d_out, int out_size, void* d_ws, size_t ws_size,
                              hipStream_t stream) {
    const float* gp   = (const float*)d_in[0];
    // d_in[1] = instance_feature_pool: unused by the reference
    const float* zmap = (const float*)d_in[2];
    const float* w2c  = (const float*)d_in[3];
    const float* camk = (const float*)d_in[4];
    const float* voxo = (const float*)d_in[5];
    const float* ssz  = (const float*)d_in[6];
    const float* cvr  = (const float*)d_in[7];
    const float* Wi   = (const float*)d_in[8];
    const float* bi   = (const float*)d_in[9];
    const float* W1   = (const float*)d_in[10];
    const float* b1   = (const float*)d_in[11];
    const float* W2   = (const float*)d_in[12];
    const float* b2   = (const float*)d_in[13];
    const float* W3   = (const float*)d_in[14];
    const float* b3   = (const float*)d_in[15];

    float* out        = (float*)d_out;
    float* out_anchor = out;                                  // N*23
    float* out_tag    = out_anchor + (size_t)N_PTS * 23;      // N
    float* out_inst   = out_tag + N_PTS;                      // N*96
    float* out_pool   = out_inst + (size_t)N_PTS * 96;        // N*25
    float* out_mask   = out_pool + (size_t)N_PTS * 25;        // N

    const size_t smemBytes = (size_t)SMEM_F16 * sizeof(_Float16); // 142336 B
    dim3 grid(N_PTS / BLOCK);                                 // 3125 (exact)
    hipLaunchKernelGGL(lifter_kernel, grid, dim3(BLOCK), smemBytes, stream,
                       gp, zmap, w2c, camk, voxo, ssz, cvr,
                       Wi, bi, W1, b1, W2, b2, W3, b3,
                       out_anchor, out_tag, out_inst, out_pool, out_mask);
}
